// RITBlock_30709016167110
// MI455X (gfx1250) — compile-verified, matrix operand order checked by reading, not run on hardware
//
#include <hip/hip_runtime.h>
#include <math.h>

// ---------------------------------------------------------------------------
// Problem constants (from reference)
// ---------------------------------------------------------------------------
#define HH    192
#define WW    192
#define HWPIX 36864            // 192*192
#define CIN   32
#define COUT  32
#define HIDN  64
#define MNN   256              // M1*M2
#define DIN   34               // 2 + CIN
#define NB    4
#define SUBPIX 64              // pixels per sub-tile (WMMA M span)
#define NSUB   4               // sub-tiles per block
#define PPB    256             // pixels per block
#define TPB    144             // tiles per batch = HWPIX / PPB

typedef __attribute__((ext_vector_type(16))) __bf16 v16bf;
typedef __attribute__((ext_vector_type(8)))  float  v8f;

#define DEV __device__ __forceinline__

DEV v8f wmma_bf16(v16bf a, v16bf b, v8f c) {
  return __builtin_amdgcn_wmma_f32_16x16x32_bf16(
      /*neg_a=*/false, a, /*neg_b=*/false, b,
      /*c_mod=*/(short)0, c, /*reuse_a=*/false, /*reuse_b=*/false);
}

// Branchless tanh-form GELU (fast v_exp_f32 path; error below the bf16
// matmul noise floor; compiler vectorizes pairs into v_pk_* ops).
DEV float gelu_f(float x) {
  const float y  = 0.7978845608028654f * (x + 0.044715f * x * x * x);
  const float t  = __expf(-2.0f * fabsf(y));      // e^{-2|y|}
  float th = (1.0f - t) / (1.0f + t);             // tanh(|y|)
  th = copysignf(th, y);
  return 0.5f * x * (1.0f + th);
}

// A-style fragment from LDS, operand stored row-major as (rows x K).
// lane<16: row=lane,   K in {k0..k0+7, k0+16..k0+23}
// lane>=16: row=lane-16, K in {k0+8..k0+15, k0+24..k0+31}
DEV v16bf frag_lds_NK(const __bf16* base, int ld, int r0, int k0) {
  const int lane = threadIdx.x & 31;
  const int r    = lane & 15;
  const int kb   = (lane & 16) ? 8 : 0;
  const __bf16* p = base + (r0 + r) * ld + k0 + kb;
  union { uint4 u[2]; v16bf v; } c;
  c.u[0] = *reinterpret_cast<const uint4*>(p);
  c.u[1] = *reinterpret_cast<const uint4*>(p + 16);
  return c.v;
}

// B fragment from LDS stored row-major as (K x N) (strided per-lane reads).
DEV v16bf frag_lds_KN(const __bf16* base, int ld, int k0, int n0) {
  const int lane = threadIdx.x & 31;
  const int n  = n0 + (lane & 15);
  const int kb = k0 + ((lane & 16) ? 8 : 0);
  v16bf v;
#pragma unroll
  for (int i = 0; i < 8; ++i) {
    v[i]     = base[(kb + i) * ld + n];
    v[i + 8] = base[(kb + 16 + i) * ld + n];
  }
  return v;
}

// Pre-packed weight fragment: frag f, lane -> 16 contiguous bf16 (32 bytes).
DEV v16bf frag_pk(const __bf16* __restrict__ P, int f) {
  const uint4* p = reinterpret_cast<const uint4*>(
      P + ((size_t)f * 32 + (threadIdx.x & 31)) * 16);
  union { uint4 u[2]; v16bf v; } c;
  c.u[0] = p[0];
  c.u[1] = p[1];
  return c.v;
}

// A fragment built directly from global x (rows = pixels, K = channels 0..31)
DEV v16bf frag_x_pix(const float* __restrict__ x, int bIdx, int pixG0) {
  const int lane = threadIdx.x & 31;
  const int pixG = pixG0 + (lane & 15);
  const int kb   = (lane & 16) ? 8 : 0;
  const size_t base = (size_t)bIdx * CIN * HWPIX;
  v16bf v;
#pragma unroll
  for (int i = 0; i < 8; ++i) {
    v[i]     = (__bf16)x[base + (size_t)(kb + i) * HWPIX + pixG];
    v[i + 8] = (__bf16)x[base + (size_t)(kb + 16 + i) * HWPIX + pixG];
  }
  return v;
}

// Scatter C/D (v8f) into row-major LDS with activation already applied.
DEV void store_c_lds(__bf16* base, int ld, int m0, int n0, const v8f& c) {
  const int lane = threadIdx.x & 31;
  const int n  = n0 + (lane & 15);
  const int mb = m0 + ((lane & 16) ? 8 : 0);
#pragma unroll
  for (int g = 0; g < 8; ++g) base[(mb + g) * ld + n] = (__bf16)c[g];
}

// ---------------------------------------------------------------------------
// Shared MLP: build 64-pixel basis tile (64 x 256) in LDS.
// sXin may alias sH1 (sXin dead after GEMM1). Force-inlined so global
// pointer args keep addrspace(1) -> global_load (not flat) codegen.
// ---------------------------------------------------------------------------
DEV void compute_basis(const float* __restrict__ x,
                       const float* __restrict__ b0,
                       const float* __restrict__ b1,
                       const float* __restrict__ b2,
                       const __bf16* __restrict__ pw0,
                       const __bf16* __restrict__ pw1,
                       const __bf16* __restrict__ pw2,
                       __bf16* sXin, __bf16* sH0, __bf16* sH1, __bf16* sBasis,
                       __bf16* sXT, int bIdx, int pixBase) {
  const int tid  = threadIdx.x;
  const int wave = tid >> 5;
  const int lane = tid & 31;

  // ---- load x tile -> x_in (64 x 64 bf16) [+ optional transposed copy] ----
#pragma unroll
  for (int it = 0; it < (SUBPIX * CIN) / 256; ++it) {
    const int idx = it * 256 + tid;
    const int pix = idx & 63;
    const int c   = idx >> 6;
    const float v = x[((size_t)(bIdx * CIN + c)) * HWPIX + pixBase + pix];
    const __bf16 bv = (__bf16)v;
    sXin[pix * 64 + 2 + c] = bv;
    if (sXT) sXT[c * 64 + pix] = bv;
  }
  if (tid < SUBPIX) {
    const int pg = pixBase + tid;
    const int hh = pg / WW;
    const int ww = pg - hh * WW;
    sXin[tid * 64 + 0] = (__bf16)(2.0f * (float)hh * (1.0f / 191.0f) - 1.0f);
    sXin[tid * 64 + 1] = (__bf16)(2.0f * (float)ww * (1.0f / 191.0f) - 1.0f);
#pragma unroll
    for (int cp = DIN; cp < 64; cp += 2)
      *reinterpret_cast<unsigned int*>(sXin + tid * 64 + cp) = 0u;
  }
  __syncthreads();

  // ---- GEMM1: (64x64) @ w0 (64x64, zero-padded K>=34) -> GELU -> sH0 ----
#pragma unroll
  for (int i = 0; i < 2; ++i) {
    const int t  = wave + i * 8;
    const int m0 = (t & 3) * 16, n0 = (t >> 2) * 16;
    v8f acc = {0.f, 0.f, 0.f, 0.f, 0.f, 0.f, 0.f, 0.f};
#pragma unroll
    for (int kk = 0; kk < 2; ++kk) {
      v16bf a  = frag_lds_NK(sXin, 64, m0, kk * 32);
      v16bf bb = frag_pk(pw0, kk * 4 + (n0 >> 4));
      acc = wmma_bf16(a, bb, acc);
    }
    const float bias = b0[n0 + (lane & 15)];
#pragma unroll
    for (int g = 0; g < 8; ++g) acc[g] = gelu_f(acc[g] + bias);
    store_c_lds(sH0, 64, m0, n0, acc);
  }
  __syncthreads();

  // ---- GEMM2: sH0 @ w1 -> GELU -> sH1 (may overwrite sXin region) ----
#pragma unroll
  for (int i = 0; i < 2; ++i) {
    const int t  = wave + i * 8;
    const int m0 = (t & 3) * 16, n0 = (t >> 2) * 16;
    v8f acc = {0.f, 0.f, 0.f, 0.f, 0.f, 0.f, 0.f, 0.f};
#pragma unroll
    for (int kk = 0; kk < 2; ++kk) {
      v16bf a  = frag_lds_NK(sH0, 64, m0, kk * 32);
      v16bf bb = frag_pk(pw1, kk * 4 + (n0 >> 4));
      acc = wmma_bf16(a, bb, acc);
    }
    const float bias = b1[n0 + (lane & 15)];
#pragma unroll
    for (int g = 0; g < 8; ++g) acc[g] = gelu_f(acc[g] + bias);
    store_c_lds(sH1, 64, m0, n0, acc);
  }
  __syncthreads();

  // ---- GEMM3: sH1 @ w2 (64x256) + b2 -> sBasis (64 x 256) ----
  // unroll capped at 4: keeps <=8 weight fragments in flight (VGPR pressure)
#pragma unroll 4
  for (int i = 0; i < 8; ++i) {
    const int t  = wave + i * 8;
    const int m0 = (t & 3) * 16, n0 = (t >> 2) * 16;
    v8f acc = {0.f, 0.f, 0.f, 0.f, 0.f, 0.f, 0.f, 0.f};
#pragma unroll
    for (int kk = 0; kk < 2; ++kk) {
      v16bf a  = frag_lds_NK(sH1, 64, m0, kk * 32);
      v16bf bb = frag_pk(pw2, kk * 16 + (n0 >> 4));
      acc = wmma_bf16(a, bb, acc);
    }
    const float bias = b2[n0 + (lane & 15)];
#pragma unroll
    for (int g = 0; g < 8; ++g) acc[g] += bias;
    store_c_lds(sBasis, 256, m0, n0, acc);
  }
  __syncthreads();
}

// ---------------------------------------------------------------------------
// Kernel: pack all weights into WMMA-fragment-native bf16 layout in ws.
// 52 fragments: w0[8] w1[8] w2[32] w_mix[2] w_sc[2]; each frag = 32 lanes x 16.
// ---------------------------------------------------------------------------
__global__ void k_pack(const float* __restrict__ w0, const float* __restrict__ w1,
                       const float* __restrict__ w2, const float* __restrict__ wmix,
                       const float* __restrict__ wsc, __bf16* __restrict__ pw) {
  const int gid = blockIdx.x * 256 + threadIdx.x;
  if (gid >= 52 * 32) return;
  const int f    = gid >> 5;
  const int lane = gid & 31;
  const int r    = lane & 15;
  const int kb8  = (lane & 16) ? 8 : 0;
  __bf16 v[16];

  if (f < 8) {                       // w_mlp0 (KxN = 64pad x 64), zero K>=34
    const int kt = f >> 2, nt = f & 3;
    const int n = nt * 16 + r, kb = kt * 32 + kb8;
#pragma unroll
    for (int i = 0; i < 8; ++i) {
      const int k1 = kb + i, k2 = kb + 16 + i;
      v[i]     = (k1 < DIN) ? (__bf16)w0[k1 * 64 + n] : (__bf16)0.0f;
      v[i + 8] = (k2 < DIN) ? (__bf16)w0[k2 * 64 + n] : (__bf16)0.0f;
    }
  } else if (f < 16) {               // w_mlp1 (64 x 64)
    const int fl = f - 8, kt = fl >> 2, nt = fl & 3;
    const int n = nt * 16 + r, kb = kt * 32 + kb8;
#pragma unroll
    for (int i = 0; i < 8; ++i) {
      v[i]     = (__bf16)w1[(kb + i) * 64 + n];
      v[i + 8] = (__bf16)w1[(kb + 16 + i) * 64 + n];
    }
  } else if (f < 48) {               // w_mlp2 (64 x 256)
    const int fl = f - 16, kt = fl >> 4, nt = fl & 15;
    const int n = nt * 16 + r, kb = kt * 32 + kb8;
#pragma unroll
    for (int i = 0; i < 8; ++i) {
      v[i]     = (__bf16)w2[(kb + i) * 256 + n];
      v[i + 8] = (__bf16)w2[(kb + 16 + i) * 256 + n];
    }
  } else if (f < 50) {               // w_mix (KxN = 32 x 32)
    const int nt = f - 48;
    const int n = nt * 16 + r;
#pragma unroll
    for (int i = 0; i < 8; ++i) {
      v[i]     = (__bf16)wmix[(kb8 + i) * 32 + n];
      v[i + 8] = (__bf16)wmix[(kb8 + 16 + i) * 32 + n];
    }
  } else {                           // w_sc stored (o,c) = B^T row-major (NxK)
    const int nt = f - 50;
    const int rr = nt * 16 + r;
#pragma unroll
    for (int i = 0; i < 8; ++i) {
      v[i]     = (__bf16)wsc[rr * 32 + kb8 + i];
      v[i + 8] = (__bf16)wsc[rr * 32 + kb8 + 16 + i];
    }
  }

  __bf16* d = pw + ((size_t)f * 32 + lane) * 16;
#pragma unroll
  for (int i = 0; i < 16; ++i) d[i] = v[i];
}

// ---------------------------------------------------------------------------
// Kernel: forward transform. Per block: 256 pixels, 4 sub-tiles.
// Writes deterministic per-block partial xhat (32 x 256 f32).
// ---------------------------------------------------------------------------
__global__ __launch_bounds__(256) void k_fwd(
    const float* __restrict__ x,
    const float* __restrict__ b0, const float* __restrict__ b1,
    const float* __restrict__ b2,
    const __bf16* __restrict__ pw0, const __bf16* __restrict__ pw1,
    const __bf16* __restrict__ pw2,
    float* __restrict__ partial) {
  __shared__ __align__(16) __bf16 smem[30720];  // 60 KB
  __bf16* sXin   = smem;              // 4096  (64 x 64)
  __bf16* sXT    = smem + 4096;       // 2048  (32 x 64) x transposed
  __bf16* sH0    = smem + 6144;       // 4096
  __bf16* sH1    = smem + 10240;      // 4096
  __bf16* sBasis = smem + 14336;      // 16384 (64 x 256)

  const int tile = blockIdx.x;
  const int b    = blockIdx.y;
  const int wave = threadIdx.x >> 5;
  const int lane = threadIdx.x & 31;

  v8f acc4[4];
#pragma unroll
  for (int j = 0; j < 4; ++j) acc4[j] = (v8f){0.f,0.f,0.f,0.f,0.f,0.f,0.f,0.f};

  const int m0w = (wave >> 2) * 16;   // channel-tile row for this wave

#pragma unroll 1
  for (int s = 0; s < NSUB; ++s) {
    // scheduling fence: stop the compiler hoisting the (L1-hot) packed-weight
    // fragment loads out of this loop -- reloading them is far cheaper than
    // keeping ~24 fragments (~190 VGPRs) live across iterations.
    asm volatile("" ::: "memory");

    const int pixBase = tile * PPB + s * SUBPIX;
    if (s < NSUB - 1)
      __builtin_prefetch(x + (size_t)b * CIN * HWPIX + pixBase + SUBPIX, 0, 0);

    compute_basis(x, b0, b1, b2, pw0, pw1, pw2,
                  sXin, sH0, sH1, sBasis, sXT, b, pixBase);

    // GEMM4: xhat partial (32 x 256) += xT (32 x 64) @ basis (64 x 256)
    v16bf a0 = frag_lds_NK(sXT, 64, m0w, 0);
    v16bf a1 = frag_lds_NK(sXT, 64, m0w, 32);
#pragma unroll
    for (int j = 0; j < 4; ++j) {
      const int n0 = (wave & 3) * 64 + j * 16;
      v16bf bb0 = frag_lds_KN(sBasis, 256, 0, n0);
      acc4[j] = wmma_bf16(a0, bb0, acc4[j]);
      v16bf bb1 = frag_lds_KN(sBasis, 256, 32, n0);
      acc4[j] = wmma_bf16(a1, bb1, acc4[j]);
    }
    __syncthreads();  // protect LDS before next sub-tile overwrites
  }

  // write per-block partial (deterministic; no atomics)
  float* pp = partial + ((size_t)(b * TPB + tile)) * (CIN * MNN);
  const int mb = (lane & 16) ? 8 : 0;
  const int nn = lane & 15;
#pragma unroll
  for (int j = 0; j < 4; ++j) {
    const int n0 = (wave & 3) * 64 + j * 16;
#pragma unroll
    for (int g = 0; g < 8; ++g)
      pp[(m0w + mb + g) * MNN + n0 + nn] = acc4[j][g];
  }
}

// ---------------------------------------------------------------------------
// Kernel: deterministic reduction of partials -> xacc (B x CIN x MN)
// ---------------------------------------------------------------------------
__global__ void k_reduce(const float* __restrict__ partial,
                         float* __restrict__ xacc) {
  const int i = blockIdx.x * 256 + threadIdx.x;   // 32768
  const int b  = i >> 13;
  const int cm = i & 8191;
  const float* p = partial + (size_t)b * TPB * 8192 + cm;
  float s = 0.f;
  for (int t = 0; t < TPB; ++t) s += p[(size_t)t * 8192];
  xacc[i] = s;
}

// ---------------------------------------------------------------------------
// Kernel: spectral weighting: xhat2[b,o,mn] = sum_c lw2[c,o]*lw1[c,mn]*xacc
// Output bf16 in B^T-fragment-friendly row-major (o x mn) layout.
// ---------------------------------------------------------------------------
__global__ void k_spec(const float* __restrict__ xacc,
                       const float* __restrict__ lw1,
                       const float* __restrict__ lw2,
                       __bf16* __restrict__ xhat2) {
  const int i  = blockIdx.x * 256 + threadIdx.x;  // 32768
  const int mn = i & 255;
  const int o  = (i >> 8) & 31;
  const int b  = i >> 13;
  float s = 0.f;
#pragma unroll
  for (int c = 0; c < CIN; ++c)
    s += lw2[c * COUT + o] * lw1[c * MNN + mn] * xacc[((b * CIN + c) << 8) + mn];
  xhat2[i] = (__bf16)s;
}

// ---------------------------------------------------------------------------
// Kernel: inverse transform + channel mixer + shortcut + GELUs.
// Recomputes basis (cheaper than 75 MB round-trip for it through HBM).
// ---------------------------------------------------------------------------
__global__ __launch_bounds__(256) void k_inv(
    const float* __restrict__ x,
    const float* __restrict__ b0, const float* __restrict__ b1,
    const float* __restrict__ b2,
    const __bf16* __restrict__ pw0, const __bf16* __restrict__ pw1,
    const __bf16* __restrict__ pw2,
    const __bf16* __restrict__ xhat2,
    const float* __restrict__ alpha,
    const float* __restrict__ b_mix,
    const __bf16* __restrict__ pwmix, const __bf16* __restrict__ pwsc,
    const float* __restrict__ b_sc,
    float* __restrict__ out) {
  __shared__ __align__(16) __bf16 smem[32768];  // 64 KB, phase-aliased
  __bf16* sXin   = smem;              // 4096 (aliases sH1; sXin dead after G1)
  __bf16* sH1    = smem;
  __bf16* sH0    = smem + 4096;       // 4096 (aliases sXrec; sH0 dead after G2)
  __bf16* sXrec  = smem + 4096;
  __bf16* sBasis = smem + 8192;       // 16384
  __bf16* sXh2   = smem + 24576;      // 8192 (32 x 256)

  const int tile = blockIdx.x;
  const int b    = blockIdx.y;
  const int wave = threadIdx.x >> 5;
  const int lane = threadIdx.x & 31;

  // load this batch's xhat2 slice (B^T layout: o rows x mn K) once
  {
    const uint4* src = reinterpret_cast<const uint4*>(xhat2 + (size_t)b * 8192);
    uint4* dst = reinterpret_cast<uint4*>(sXh2);
#pragma unroll
    for (int it = 0; it < 4; ++it) dst[it * 256 + threadIdx.x] = src[it * 256 + threadIdx.x];
  }

  const int m0  = (wave & 3) * 16;    // pixel tile row
  const int n0c = (wave >> 2) * 16;   // channel tile col
  const int nn  = lane & 15;
  const int mb  = (lane & 16) ? 8 : 0;

#pragma unroll 1
  for (int s = 0; s < NSUB; ++s) {
    // scheduling fence: see k_fwd (register-pressure valve)
    asm volatile("" ::: "memory");

    const int pixBase = tile * PPB + s * SUBPIX;

    compute_basis(x, b0, b1, b2, pw0, pw1, pw2,
                  sXin, sH0, sH1, sBasis, nullptr, b, pixBase);
    // (compute_basis ends with a barrier; sXh2 load is visible by now)

    // GEMM5: x_rec (64 x 32) = basis (64 x 256) @ xhat2^T (256 x 32)
    v8f a5 = {0.f, 0.f, 0.f, 0.f, 0.f, 0.f, 0.f, 0.f};
#pragma unroll
    for (int k = 0; k < MNN; k += 32) {
      v16bf a  = frag_lds_NK(sBasis, 256, m0, k);
      v16bf bb = frag_lds_NK(sXh2, 256, n0c, k);
      a5 = wmma_bf16(a, bb, a5);
    }
    const float al = alpha[n0c + nn];
#pragma unroll
    for (int g = 0; g < 8; ++g) a5[g] *= al;
    store_c_lds(sXrec, 32, m0, n0c, a5);
    __syncthreads();

    // GEMM6: mixer  out6 = x_rec @ w_mix (+b_mix, GELU)
    v8f a6 = {0.f, 0.f, 0.f, 0.f, 0.f, 0.f, 0.f, 0.f};
    {
      v16bf a  = frag_lds_NK(sXrec, 32, m0, 0);
      v16bf bb = frag_pk(pwmix, n0c >> 4);
      a6 = wmma_bf16(a, bb, a6);
    }
    const float bm = b_mix[n0c + nn];

    // GEMM7: shortcut = x @ w_sc^T (+b_sc)
    v8f a7 = {0.f, 0.f, 0.f, 0.f, 0.f, 0.f, 0.f, 0.f};
    {
      v16bf a  = frag_x_pix(x, b, pixBase + m0);
      v16bf bb = frag_pk(pwsc, n0c >> 4);
      a7 = wmma_bf16(a, bb, a7);
    }
    const float bs = b_sc[n0c + nn];

    // epilogue: gelu(gelu(mix) + shortcut), packed stores
    const int ch = n0c + nn;
    float fr[8];
#pragma unroll
    for (int g = 0; g < 8; ++g) {
      const float u = gelu_f(a6[g] + bm);
      fr[g] = gelu_f(u + a7[g] + bs);
    }
    float* op = out + ((size_t)(b * COUT + ch)) * HWPIX + pixBase + m0 + mb;
    *reinterpret_cast<float4*>(op)     = make_float4(fr[0], fr[1], fr[2], fr[3]);
    *reinterpret_cast<float4*>(op + 4) = make_float4(fr[4], fr[5], fr[6], fr[7]);

    __syncthreads();  // sXrec consumed by all waves before next sub-tile
  }
}

// ---------------------------------------------------------------------------
// Host launcher
// ---------------------------------------------------------------------------
extern "C" void kernel_launch(void* const* d_in, const int* in_sizes, int n_in,
                              void* d_out, int out_size, void* d_ws, size_t ws_size,
                              hipStream_t stream) {
  const float* x      = (const float*)d_in[0];
  const float* w_mlp0 = (const float*)d_in[1];
  const float* b_mlp0 = (const float*)d_in[2];
  const float* w_mlp1 = (const float*)d_in[3];
  const float* b_mlp1 = (const float*)d_in[4];
  const float* w_mlp2 = (const float*)d_in[5];
  const float* b_mlp2 = (const float*)d_in[6];
  const float* lw1    = (const float*)d_in[7];
  const float* lw2    = (const float*)d_in[8];
  const float* alpha  = (const float*)d_in[9];
  const float* w_mix  = (const float*)d_in[10];
  const float* b_mix  = (const float*)d_in[11];
  const float* w_sc   = (const float*)d_in[12];
  const float* b_sc   = (const float*)d_in[13];
  float* out = (float*)d_out;

  // workspace layout (bytes):
  //   [0)                partial  : B*TPB * 32*256 f32 = 18,874,368
  //   [18874368)         xacc     : B*32*256 f32       = 131,072
  //   [19005440)         xhat2    : B*32*256 bf16      = 65,536
  //   [19070976)         packed W : 52*512 bf16        = 53,248
  char* wsb = (char*)d_ws;
  float*  partial = (float*) (wsb);
  float*  xacc    = (float*) (wsb + 18874368);
  __bf16* xhat2   = (__bf16*)(wsb + 19005440);
  __bf16* pw      = (__bf16*)(wsb + 19070976);
  __bf16* pw0   = pw;            // 8 frags
  __bf16* pw1   = pw + 4096;     // 8 frags
  __bf16* pw2   = pw + 8192;     // 32 frags
  __bf16* pwmix = pw + 24576;    // 2 frags
  __bf16* pwsc  = pw + 25600;    // 2 frags

  k_pack<<<7, 256, 0, stream>>>(w_mlp0, w_mlp1, w_mlp2, w_mix, w_sc, pw);

  dim3 gridF(TPB, NB);
  k_fwd<<<gridF, 256, 0, stream>>>(x, b_mlp0, b_mlp1, b_mlp2,
                                   pw0, pw1, pw2, partial);

  k_reduce<<<128, 256, 0, stream>>>(partial, xacc);
  k_spec<<<128, 256, 0, stream>>>(xacc, lw1, lw2, xhat2);

  k_inv<<<gridF, 256, 0, stream>>>(x, b_mlp0, b_mlp1, b_mlp2,
                                   pw0, pw1, pw2, xhat2, alpha,
                                   b_mix, pwmix, pwsc, b_sc, out);
}
